// Gate_10136122819135
// MI455X (gfx1250) — matrix operand order repaired and checked by reading, not compiled, not run on hardware
//
#include <hip/hip_runtime.h>
#include <hip/hip_bf16.h>

// MoE gate / router for MI455X (gfx1250).
//   scores = x @ W^T + b ; softmax ; top-8 ; usage histogram.
// Roofline: 34.4 GFLOP vs ~1.1 GB traffic (x streamed once) -> target is the
// 23.3 TB/s HBM floor (~47 us). GEMM must run on the bf16 WMMA pipe
// (v_wmma_f32_16x16x32_bf16, f32 accumulate) to stay under that floor.
// Data path: x is DMA'd to LDS with CDNA5 async loads (ASYNCcnt, no VGPR
// round trip); f32->bf16 conversion happens in the A-fragment build
// (v_cvt_pk_bf16_f32), amortized over the 4 N-tiles.

#define DIM        1024
#define NEXP       64
#define TOPKK      8
#define BATCH      262144
#define TPB        128          // 4 waves (wave32)
#define TOKPB      128          // tokens per block
#define KC         32           // K chunk per LDS stage
#define XSF_STRIDE 36           // f32 elems per x row in LDS (32 + 4 pad, 16B aligned)
#define WS_STRIDE  40           // bf16 elems per W row in LDS (32 + 8 pad)
#define SC_STRIDE  65           // f32 elems per score row in LDS (64 + 1 pad)

typedef __attribute__((ext_vector_type(16))) __bf16 v16bf;
typedef __attribute__((ext_vector_type(8)))  __bf16 v8bf;
typedef __attribute__((ext_vector_type(4)))  __bf16 v4bf;
typedef __attribute__((ext_vector_type(8)))  float  v8f;
typedef __attribute__((ext_vector_type(4)))  float  v4f;

__global__ __launch_bounds__(TPB)
void gate_init_kernel(float* __restrict__ tail) {
    // tail points at d_out[2*B*TOPK]: 64 histogram bins + total_tokens.
    int t = threadIdx.x;
    if (t < NEXP) tail[t] = 0.0f;
    if (t == NEXP) tail[NEXP] = (float)((unsigned)BATCH * (unsigned)TOPKK); // 2097152
}

__device__ __forceinline__ v4bf cvt4(v4f v) {
    return (v4bf){ (__bf16)v[0], (__bf16)v[1], (__bf16)v[2], (__bf16)v[3] };
}

__global__ __launch_bounds__(TPB)
void gate_main_kernel(const float* __restrict__ x,
                      const float* __restrict__ Wm,
                      const float* __restrict__ bias,
                      float* __restrict__ out) {
    __shared__ __align__(16) float  xsf[TOKPB * XSF_STRIDE];   // x chunk, raw f32 via async DMA
    __shared__ __align__(16) __bf16 wsh[NEXP * WS_STRIDE];     // W chunk, pre-converted bf16
    __shared__ __align__(16) float  sc[TOKPB * SC_STRIDE];     // score tile for epilogue
    __shared__ float    bsh[NEXP];
    __shared__ unsigned hist[NEXP];

    const int tid   = threadIdx.x;
    const int wave  = tid >> 5;
    const int lane  = tid & 31;
    const int khalf = lane >> 4;      // K-half selector for A/B fragments
    const int l16   = lane & 15;
    const size_t tok_base = (size_t)blockIdx.x * TOKPB;

    if (tid < NEXP) { bsh[tid] = bias[tid]; hist[tid] = 0u; }

    v8f acc[2][4] = {};               // 32 tokens x 64 experts per wave

    const int sub  = tid & 7;         // 16B slot within a 32-wide row
    const int rsel = tid >> 3;        // 0..15: row-within-pass

    for (int kc = 0; kc < DIM; kc += KC) {
        __syncthreads();              // previous chunk fully consumed by all waves

        // ---- stage x tile [128 x 32] f32 straight to LDS: async DMA, no VGPR round trip
        #pragma unroll
        for (int p = 0; p < 8; ++p) {
            const int row = p * 16 + rsel;
            const float* gp = x + tok_base * DIM + (size_t)row * DIM + kc + sub * 4;
            unsigned lds_off = (unsigned)(uintptr_t)&xsf[row * XSF_STRIDE + sub * 4];
            asm volatile("global_load_async_to_lds_b128 %0, %1, off th:TH_LOAD_NT"
                         :: "v"(lds_off), "v"(gp) : "memory");
        }
        // ---- stage W tile [64 x 32] f32 -> bf16 LDS (tiny, L2-resident)
        #pragma unroll
        for (int p = 0; p < 4; ++p) {
            const int row = p * 16 + rsel;
            const v4f v = *(const v4f*)(Wm + (size_t)row * DIM + kc + sub * 4);
            *(v4bf*)(&wsh[row * WS_STRIDE + sub * 4]) = cvt4(v);
        }
        asm volatile("s_wait_asynccnt 0x0" ::: "memory");  // my async writes landed in LDS
        __syncthreads();                                   // everyone's writes visible

        // ---- B fragments: lane L holds N = nt*16 + (L&15), K = khalf*8 + e (e<8), 16 + khalf*8 + (e-8) (e>=8)
        //      (two 8-wide bf16 runs of one W row -> two ds_load_b128)
        union { v16bf v; v8bf h[2]; } bfr[4];
        #pragma unroll
        for (int nt = 0; nt < 4; ++nt) {
            const int brow = nt * 16 + l16;
            bfr[nt].h[0] = *(const v8bf*)(&wsh[brow * WS_STRIDE + khalf * 8]);
            bfr[nt].h[1] = *(const v8bf*)(&wsh[brow * WS_STRIDE + khalf * 8 + 16]);
        }
        // ---- A fragments (f32 ds_load_b128 + packed bf16 convert) + 8 WMMAs
        #pragma unroll
        for (int mt = 0; mt < 2; ++mt) {
            const int arow = wave * 32 + mt * 16 + l16;   // lanes 0-15/16-31: same M, K halves
            const float* ap = &xsf[arow * XSF_STRIDE + khalf * 8];
            const v4f f0 = *(const v4f*)(ap + 0);         // K = khalf*8 + 0..3
            const v4f f1 = *(const v4f*)(ap + 4);         // K = khalf*8 + 4..7
            const v4f f2 = *(const v4f*)(ap + 16);        // K = 16 + khalf*8 + 0..3
            const v4f f3 = *(const v4f*)(ap + 20);        // K = 16 + khalf*8 + 4..7
            union { v16bf v; v4bf q[4]; } afr;
            afr.q[0] = cvt4(f0);
            afr.q[1] = cvt4(f1);
            afr.q[2] = cvt4(f2);
            afr.q[3] = cvt4(f3);
            #pragma unroll
            for (int nt = 0; nt < 4; ++nt) {
                acc[mt][nt] = __builtin_amdgcn_wmma_f32_16x16x32_bf16(
                    false, afr.v, false, bfr[nt].v, (short)0, acc[mt][nt], false, false);
            }
        }
    }

    // ---- scatter scores to LDS per documented C/D layout:
    //      lanes 0-15: M = r ; lanes 16-31: M = r + 8 ; N = lane & 15
    #pragma unroll
    for (int mt = 0; mt < 2; ++mt)
        #pragma unroll
        for (int nt = 0; nt < 4; ++nt)
            #pragma unroll
            for (int r = 0; r < 8; ++r) {
                const int tr = wave * 32 + mt * 16 + khalf * 8 + r;
                const int e  = nt * 16 + l16;
                sc[tr * SC_STRIDE + e] = acc[mt][nt][r];
            }
    __syncthreads();

    // ---- one token per thread: bias, softmax stats, top-8 (register-resident)
    float s[NEXP];
    #pragma unroll
    for (int e = 0; e < NEXP; ++e) s[e] = sc[tid * SC_STRIDE + e] + bsh[e];

    float mx = s[0];
    #pragma unroll
    for (int e = 1; e < NEXP; ++e) mx = fmaxf(mx, s[e]);
    float sum = 0.0f;
    #pragma unroll
    for (int e = 0; e < NEXP; ++e) sum += __expf(s[e] - mx);
    const float inv = __builtin_amdgcn_rcpf(sum);

    const size_t token = tok_base + tid;
    float* __restrict__ w_out = out;
    float* __restrict__ i_out = out + (size_t)BATCH * TOPKK;

    unsigned long long taken = 0ull;
    #pragma unroll
    for (int j = 0; j < TOPKK; ++j) {
        float best = -3.402823466e38f;
        int   bi   = 0;
        #pragma unroll
        for (int e = 0; e < NEXP; ++e) {
            const bool free_e = ((taken >> e) & 1ull) == 0ull;
            if (free_e && s[e] > best) { best = s[e]; bi = e; } // ties -> lowest index (top_k order)
        }
        taken |= (1ull << bi);
        __builtin_nontemporal_store(__expf(best - mx) * inv, &w_out[token * TOPKK + j]);
        __builtin_nontemporal_store((float)bi,               &i_out[token * TOPKK + j]);
        atomicAdd(&hist[bi], 1u);                 // ds_add_u32
    }
    __syncthreads();

    if (tid < NEXP)                               // counts < 2^24 -> exact, order-independent
        atomicAdd(out + (size_t)BATCH * TOPKK * 2 + tid, (float)hist[tid]);
}

extern "C" void kernel_launch(void* const* d_in, const int* in_sizes, int n_in,
                              void* d_out, int out_size, void* d_ws, size_t ws_size,
                              hipStream_t stream) {
    const float* x  = (const float*)d_in[0];   // [B, DIM]
    const float* Wm = (const float*)d_in[1];   // [NEXP, DIM]
    const float* b  = (const float*)d_in[2];   // [NEXP]
    float* out = (float*)d_out;                // weights | indices(as f32) | hist | total

    hipLaunchKernelGGL(gate_init_kernel, dim3(1), dim3(TPB), 0, stream,
                       out + (size_t)BATCH * TOPKK * 2);
    hipLaunchKernelGGL(gate_main_kernel, dim3(BATCH / TOKPB), dim3(TPB), 0, stream,
                       x, Wm, b, out);
}
